// EquivariantUpdate_38431367365235
// MI455X (gfx1250) — matrix operand order repaired and checked by reading, not compiled, not run on hardware
//
#include <hip/hip_runtime.h>
#include <hip/hip_bf16.h>

#define N_NODES 16384
#define N_EDGES 65536
#define HDIM 128

typedef __attribute__((ext_vector_type(16))) __bf16 v16bf;
typedef __attribute__((ext_vector_type(8)))  __bf16 v8bf;
typedef __attribute__((ext_vector_type(8)))  float  v8f;
typedef __attribute__((ext_vector_type(4)))  unsigned int u32x4;
typedef __attribute__((ext_vector_type(8)))  int i32x8;
typedef __attribute__((ext_vector_type(4)))  int i32x4;

// ---------------------------------------------------------------------------
// WMMA fragment loaders (wave32, 16x16x32 bf16)
// ---------------------------------------------------------------------------
static __device__ __forceinline__ v16bf load_A_frag(const __bf16* act, int Kpad,
                                                    int kt, int lane) {
  int m    = lane & 15;
  int base = (lane < 16) ? 0 : 8;
  const __bf16* row = act + (size_t)m * Kpad + kt * 32;
  v8bf lo = *reinterpret_cast<const v8bf*>(row + base);
  v8bf hi = *reinterpret_cast<const v8bf*>(row + 16 + base);
  v16bf a;
#pragma unroll
  for (int x = 0; x < 8; ++x) { a[x] = lo[x]; a[8 + x] = hi[x]; }
  return a;
}

static __device__ __forceinline__ v16bf load_B_frag(const __bf16* wt, int Kpad,
                                                    int nt, int kt, int lane) {
  int n    = lane & 15;
  int koff = kt * 32 + ((lane < 16) ? 0 : 8);
  const __bf16* col = wt + (size_t)(nt * 16 + n) * Kpad + koff;
  v8bf lo = *reinterpret_cast<const v8bf*>(col);
  v8bf hi = *reinterpret_cast<const v8bf*>(col + 16);
  v16bf b;
#pragma unroll
  for (int x = 0; x < 8; ++x) { b[x] = lo[x]; b[8 + x] = hi[x]; }
  return b;
}

// GEMM with software-pipelined B fragments: B[n+1] is issued before the WMMA
// that consumes B[n], so the scheduler can hold loads in flight instead of
// draining loadcnt to 0 before every v_wmma.
template <int KT, int NT>
static __device__ __forceinline__ void gemm_bf16(const __bf16* in, const __bf16* wt,
                                                 const float* bias, __bf16* out,
                                                 int lane, bool silu) {
  v8f acc[NT];
#pragma unroll
  for (int n = 0; n < NT; ++n)
#pragma unroll
    for (int r = 0; r < 8; ++r) acc[n][r] = 0.f;

#pragma unroll
  for (int k = 0; k < KT; ++k) {
    v16bf a = load_A_frag(in, KT * 32, k, lane);
    if (k + 1 < KT) __builtin_prefetch(wt + (size_t)(k + 1) * 32, 0, 3);
    v16bf bcur = load_B_frag(wt, KT * 32, 0, k, lane);
#pragma unroll
    for (int n = 0; n < NT; ++n) {
      v16bf bnext;
      if (n + 1 < NT) bnext = load_B_frag(wt, KT * 32, n + 1, k, lane);
      acc[n] = __builtin_amdgcn_wmma_f32_16x16x32_bf16(
          false, a, false, bcur, (short)0, acc[n], false, false);
      if (n + 1 < NT) bcur = bnext;
    }
  }
  int ncol  = lane & 15;
  int mbase = (lane < 16) ? 0 : 8;
#pragma unroll
  for (int n = 0; n < NT; ++n) {
    float bv = bias[n * 16 + ncol];
#pragma unroll
    for (int r = 0; r < 8; ++r) {
      float x = acc[n][r] + bv;
      if (silu) x = x / (1.f + __expf(-x));
      out[(size_t)(mbase + r) * (NT * 16) + n * 16 + ncol] = (__bf16)x;
    }
  }
}

// Same but f32 output (pre-residual, no activation)
template <int KT, int NT>
static __device__ __forceinline__ void gemm_f32(const __bf16* in, const __bf16* wt,
                                                const float* bias, float* out, int lane) {
  v8f acc[NT];
#pragma unroll
  for (int n = 0; n < NT; ++n)
#pragma unroll
    for (int r = 0; r < 8; ++r) acc[n][r] = 0.f;

#pragma unroll
  for (int k = 0; k < KT; ++k) {
    v16bf a = load_A_frag(in, KT * 32, k, lane);
    v16bf bcur = load_B_frag(wt, KT * 32, 0, k, lane);
#pragma unroll
    for (int n = 0; n < NT; ++n) {
      v16bf bnext;
      if (n + 1 < NT) bnext = load_B_frag(wt, KT * 32, n + 1, k, lane);
      acc[n] = __builtin_amdgcn_wmma_f32_16x16x32_bf16(
          false, a, false, bcur, (short)0, acc[n], false, false);
      if (n + 1 < NT) bcur = bnext;
    }
  }
  int ncol  = lane & 15;
  int mbase = (lane < 16) ? 0 : 8;
#pragma unroll
  for (int n = 0; n < NT; ++n) {
    float bv = bias[n * 16 + ncol];
#pragma unroll
    for (int r = 0; r < 8; ++r)
      out[(size_t)(mbase + r) * (NT * 16) + n * 16 + ncol] = acc[n][r] + bv;
  }
}

// ---------------------------------------------------------------------------
// Tensor Data Mover: 1D (rows=1) copy of nelem 2-byte elements into LDS.
// D# built per CDNA5 ISA ch.8: group0 {count=1, lds_addr, global_addr, type=2},
// group1 {data_size=2B, tensor_dim0=nelem, tile_dim0=nelem, tile_dim1=1,
//         tensor_dim0_stride=nelem}. Groups 2/3 unused (zero).
// ---------------------------------------------------------------------------
static __device__ __forceinline__ void tdm_load_to_lds_2B(const void* gsrc,
                                                          void* lds_dst,
                                                          unsigned nelem) {
  unsigned lds_addr = (unsigned)(uintptr_t)lds_dst;  // low 32 bits = LDS offset
  unsigned long long ga = (unsigned long long)(uintptr_t)gsrc;
  u32x4 g0;
  g0[0] = 1u;                                            // count=1
  g0[1] = lds_addr;                                      // lds_addr [63:32]
  g0[2] = (unsigned)(ga & 0xffffffffull);                // global_addr [95:64]
  g0[3] = (unsigned)((ga >> 32) & 0x1ffffffull)          // global_addr [120:96]
          | (2u << 30);                                  // type=2 (image) [127:126]
  i32x8 g1;
  g1[0] = (int)(1u << 16);                               // data_size=1 (2 bytes)
  g1[1] = (int)((nelem & 0xffffu) << 16);                // tensor_dim0 lo16 [63:48]
  g1[2] = (int)(((nelem >> 16) & 0xffffu) | (1u << 16)); // dim0 hi16 | tensor_dim1=1
  g1[3] = (int)((nelem & 0xffffu) << 16);                // tile_dim0 [127:112]
  g1[4] = 1;                                             // tile_dim1=1 [143:128]
  g1[5] = (int)nelem;                                    // tensor_dim0_stride lo32
  g1[6] = 0;
  g1[7] = 0;
  i32x4 z4;
  z4[0] = z4[1] = z4[2] = z4[3] = 0;
#if defined(__clang_major__) && (__clang_major__ >= 23)
  i32x8 z8;
#pragma unroll
  for (int i = 0; i < 8; ++i) z8[i] = 0;
  __builtin_amdgcn_tensor_load_to_lds(g0, g1, z4, z4, z8, 0);
#else
  __builtin_amdgcn_tensor_load_to_lds(g0, g1, z4, z4, 0);
#endif
}

// ---------------------------------------------------------------------------
// Pack fp32 weights [T][K][N] -> bf16 transposed panels [T][N][Kpad] (zero pad)
// ---------------------------------------------------------------------------
__global__ void pack_transpose_kernel(const float* __restrict__ W, __bf16* __restrict__ out,
                                      int T, int K, int N, int Kpad) {
  int idx   = blockIdx.x * blockDim.x + threadIdx.x;
  int total = T * N * Kpad;
  if (idx >= total) return;
  int k = idx % Kpad;
  int n = (idx / Kpad) % N;
  int t = idx / (Kpad * N);
  float v = (k < K) ? W[((size_t)t * K + k) * N + n] : 0.f;
  out[idx] = (__bf16)v;
}

// ---------------------------------------------------------------------------
// Edge MLP: one wave per 16-edge tile; both waves of a block share one type t
// so the TDM-staged layer-3 weight panel in LDS is shared.
// 257->256->128->64->1, SiLU.
// ---------------------------------------------------------------------------
__global__ __launch_bounds__(64) void edge_mlp_kernel(
    const float* __restrict__ h, const int* __restrict__ edge_index,
    const float* __restrict__ edge_attr,
    const __bf16* __restrict__ W1b, const float* __restrict__ b1,
    const __bf16* __restrict__ W2b, const float* __restrict__ b2,
    const __bf16* __restrict__ W3b, const float* __restrict__ b3,
    const float* __restrict__ W4, const float* __restrict__ b4,
    float* __restrict__ edge_inv) {
  __shared__ __bf16 smem[2][2][16 * 288];  // 2 waves x ping-pong x [16][<=288]
  __shared__ __bf16 smemW3[64 * 128];      // TDM-staged layer-3 panel (16 KB)
  int lane  = threadIdx.x & 31;
  int wave  = threadIdx.x >> 5;
  int t     = blockIdx.x >> 11;                      // type 0..3 (uniform per block)
  int etile = (blockIdx.x & 2047) * 2 + wave;        // edge tile 0..4095
  __bf16* bufA = smem[wave][0];
  __bf16* bufB = smem[wave][1];

  // Kick off async TDM copy of the layer-3 weight panel while layers 1-2 run.
  if (wave == 0) tdm_load_to_lds_2B(W3b + (size_t)t * 64 * 128, smemW3, 64 * 128);

  // gather [h_i | h_j | ea | 0-pad] -> bufA [16][288] bf16
  for (int e = 0; e < 16; ++e) {
    int ei = etile * 16 + e;
    int vi = edge_index[ei];
    int vj = edge_index[N_EDGES + ei];
    const float* hi = h + ((size_t)vi * 4 + t) * HDIM;
    const float* hj = h + ((size_t)vj * 4 + t) * HDIM;
    for (int c = lane; c < HDIM; c += 32) {
      bufA[e * 288 + c]        = (__bf16)hi[c];
      bufA[e * 288 + HDIM + c] = (__bf16)hj[c];
    }
    if (lane == 0) bufA[e * 288 + 256] = (__bf16)edge_attr[ei];
    for (int c = 257 + lane; c < 288; c += 32) bufA[e * 288 + c] = (__bf16)0.f;
  }
  __syncthreads();
  gemm_bf16<9, 16>(bufA, W1b + (size_t)t * 256 * 288, b1 + t * 256, bufB, lane, true);
  __syncthreads();
  gemm_bf16<8, 8>(bufB, W2b + (size_t)t * 128 * 256, b2 + t * 128, bufA, lane, true);
  // Fence the TDM transfer, then barrier so both waves see the LDS panel.
  if (wave == 0) __builtin_amdgcn_s_wait_tensorcnt(0);
  __syncthreads();
  gemm_bf16<4, 4>(bufA, smemW3, b3 + t * 64, bufB, lane, true);
  __syncthreads();
  if (lane < 16) {
    const float* w4 = W4 + t * 64;
    float s = 0.f;
    for (int k = 0; k < 64; ++k) s += (float)bufB[lane * 64 + k] * w4[k];
    edge_inv[(size_t)t * N_EDGES + etile * 16 + lane] = s + b4[t];
  }
}

// ---------------------------------------------------------------------------
// Scatter: masked eq_transform atomics + scatter-mean accumulators
// ---------------------------------------------------------------------------
__global__ void scatter_kernel(const int* __restrict__ edge_index,
                               const float* __restrict__ coord,
                               const float* __restrict__ edge_attr,
                               const float* __restrict__ edge_len,
                               const unsigned char* __restrict__ edge_mask,
                               const float* __restrict__ edge_inv,
                               float* __restrict__ node_eq,
                               float* __restrict__ acc_i, float* __restrict__ cnt_i,
                               float* __restrict__ acc_j, float* __restrict__ cnt_j) {
  int e = blockIdx.x * blockDim.x + threadIdx.x;
  if (e >= N_EDGES) return;
  int vi = edge_index[e], vj = edge_index[N_EDGES + e];
  float il  = 1.f / edge_len[e];
  float dd0 = (coord[vi * 3 + 0] - coord[vj * 3 + 0]) * il;
  float dd1 = (coord[vi * 3 + 1] - coord[vj * 3 + 1]) * il;
  float dd2 = (coord[vi * 3 + 2] - coord[vj * 3 + 2]) * il;
  float ea  = edge_attr[e];
#pragma unroll
  for (int t = 0; t < 4; ++t) {
    float m = edge_mask[(size_t)t * N_EDGES + e] ? 1.f : 0.f;
    float s = edge_inv[(size_t)t * N_EDGES + e] * m;
    float* ni = node_eq + ((size_t)t * N_NODES + vi) * 3;
    float* nj = node_eq + ((size_t)t * N_NODES + vj) * 3;
    atomicAdd(ni + 0, dd0 * s);
    atomicAdd(ni + 1, dd1 * s);
    atomicAdd(ni + 2, dd2 * s);
    atomicAdd(nj + 0, -dd0 * s);
    atomicAdd(nj + 1, -dd1 * s);
    atomicAdd(nj + 2, -dd2 * s);
    atomicAdd(acc_i + t * N_NODES + vi, ea * m);
    atomicAdd(cnt_i + t * N_NODES + vi, m);
    atomicAdd(acc_j + t * N_NODES + vj, ea * m);
    atomicAdd(cnt_j + t * N_NODES + vj, m);
  }
}

// ---------------------------------------------------------------------------
// Attention: one wave per 4 nodes (16 rows = node x token). WMMA GEMMs for
// in-proj / QKV / O-proj; VALU for 4x4 softmax, LN, logits.
// ---------------------------------------------------------------------------
__global__ __launch_bounds__(64) void attn_kernel(
    const float* __restrict__ h,
    const float* __restrict__ acc_i, const float* __restrict__ cnt_i,
    const float* __restrict__ acc_j, const float* __restrict__ cnt_j,
    const __bf16* __restrict__ inWb, const float* __restrict__ in_b,
    const __bf16* __restrict__ qkvWb, const float* __restrict__ qkv_b,
    const __bf16* __restrict__ oWb, const float* __restrict__ o_b,
    const float* __restrict__ ln_g, const float* __restrict__ ln_b,
    const float* __restrict__ outW, const float* __restrict__ outb,
    float* __restrict__ logits) {
  __shared__ __align__(16) unsigned char smem_raw[2][29696];
  int lane = threadIdx.x & 31;
  int wave = threadIdx.x >> 5;
  int n0   = (blockIdx.x * 2 + wave) * 4;

  __bf16* xa  = (__bf16*)(smem_raw[wave]);           // [16][160]; reused as o [16][128]
  __bf16* x   = (__bf16*)(smem_raw[wave] + 5120);    // [16][128] xattn
  __bf16* qkv = (__bf16*)(smem_raw[wave] + 9216);    // [3][16][128]
  float*  tmp = (float*)(smem_raw[wave] + 21504);    // [16][128]

  // build w = [h | mean_i | mean_j | pad] -> xa [16][160]
  for (int idx = lane; idx < 16 * 160; idx += 32) {
    int r = idx / 160, c = idx % 160;
    int node = n0 + (r >> 2);
    int tk   = r & 3;
    float v;
    if (c < 128) {
      v = h[((size_t)node * 4 + tk) * HDIM + c];
    } else if (c == 128) {
      float cc = cnt_i[tk * N_NODES + node];
      v = acc_i[tk * N_NODES + node] / fmaxf(cc, 1.f);
    } else if (c == 129) {
      float cc = cnt_j[tk * N_NODES + node];
      v = acc_j[tk * N_NODES + node] / fmaxf(cc, 1.f);
    } else {
      v = 0.f;
    }
    xa[idx] = (__bf16)v;
  }
  __syncthreads();
  gemm_bf16<5, 8>(xa, inWb, in_b, x, lane, false);
  __syncthreads();

  for (int l = 0; l < 2; ++l) {
    for (int m = 0; m < 3; ++m)
      gemm_bf16<4, 8>(x, qkvWb + (size_t)(l * 3 + m) * 128 * 128,
                      qkv_b + (l * 3 + m) * 128, qkv + m * 2048, lane, false);
    __syncthreads();

    if (lane < 16) {  // lane = (local node, head)
      int nd = lane >> 2, hd = lane & 3;
      const __bf16* Q = qkv;
      const __bf16* K = qkv + 2048;
      const __bf16* V = qkv + 4096;
      const float scale = 0.17677669529663687f;  // 1/sqrt(32)
      for (int qt = 0; qt < 4; ++qt) {
        int qr = nd * 4 + qt;
        float sc[4];
        float mx = -1e30f;
        for (int kt = 0; kt < 4; ++kt) {
          float s = 0.f;
          for (int d = 0; d < 32; ++d)
            s += (float)Q[qr * 128 + hd * 32 + d] *
                 (float)K[(nd * 4 + kt) * 128 + hd * 32 + d];
          sc[kt] = s * scale;
          mx = fmaxf(mx, sc[kt]);
        }
        float se = 0.f;
        for (int kt = 0; kt < 4; ++kt) { sc[kt] = __expf(sc[kt] - mx); se += sc[kt]; }
        float inv = 1.f / se;
        for (int d = 0; d < 32; ++d) {
          float ov = 0.f;
          for (int kt = 0; kt < 4; ++kt)
            ov += sc[kt] * (float)V[(nd * 4 + kt) * 128 + hd * 32 + d];
          xa[qr * 128 + hd * 32 + d] = (__bf16)(ov * inv);
        }
      }
    }
    __syncthreads();
    gemm_f32<4, 8>(xa, oWb + (size_t)l * 128 * 128, o_b + l * 128, tmp, lane);
    __syncthreads();

    if (lane < 16) {  // residual + LayerNorm, one row per lane
      float mu = 0.f;
      for (int c = 0; c < 128; ++c) mu += (float)x[lane * 128 + c] + tmp[lane * 128 + c];
      mu *= (1.f / 128.f);
      float var = 0.f;
      for (int c = 0; c < 128; ++c) {
        float d = (float)x[lane * 128 + c] + tmp[lane * 128 + c] - mu;
        var += d * d;
      }
      var *= (1.f / 128.f);
      float rstd = rsqrtf(var + 1e-5f);
      for (int c = 0; c < 128; ++c) {
        float y = (float)x[lane * 128 + c] + tmp[lane * 128 + c];
        x[lane * 128 + c] = (__bf16)((y - mu) * rstd * ln_g[l * 128 + c] + ln_b[l * 128 + c]);
      }
    }
    __syncthreads();
  }

  if (lane < 16) {
    float s = 0.f;
    for (int c = 0; c < 128; ++c) s += (float)x[lane * 128 + c] * outW[c];
    s += outb[0];
    int node = n0 + (lane >> 2), tk = lane & 3;
    logits[node * 4 + tk] = s;
  }
}

// ---------------------------------------------------------------------------
// Final: softmax over 4 type logits, weighted node_eq combine, write outputs
// ---------------------------------------------------------------------------
__global__ void final_kernel(const float* __restrict__ coord,
                             const float* __restrict__ logits,
                             const float* __restrict__ node_eq,
                             float* __restrict__ out) {
  int n = blockIdx.x * blockDim.x + threadIdx.x;
  if (n >= N_NODES) return;
  float l0 = logits[n * 4 + 0], l1 = logits[n * 4 + 1];
  float l2 = logits[n * 4 + 2], l3 = logits[n * 4 + 3];
  float m  = fmaxf(fmaxf(l0, l1), fmaxf(l2, l3));
  float e0 = __expf(l0 - m), e1 = __expf(l1 - m);
  float e2 = __expf(l2 - m), e3 = __expf(l3 - m);
  float inv = 1.f / (e0 + e1 + e2 + e3);
  float w[4] = {e0 * inv, e1 * inv, e2 * inv, e3 * inv};
#pragma unroll
  for (int c = 0; c < 3; ++c) {
    float p = 0.f;
#pragma unroll
    for (int t = 0; t < 4; ++t) p += w[t] * node_eq[((size_t)t * N_NODES + n) * 3 + c];
    out[n * 3 + c] = coord[n * 3 + c] + p;
    out[(size_t)N_NODES * 3 + n * 3 + c] = p;
  }
}

// ---------------------------------------------------------------------------
extern "C" void kernel_launch(void* const* d_in, const int* in_sizes, int n_in,
                              void* d_out, int out_size, void* d_ws, size_t ws_size,
                              hipStream_t stream) {
  (void)in_sizes; (void)n_in; (void)out_size; (void)ws_size;
  const float* h          = (const float*)d_in[0];
  const float* coord      = (const float*)d_in[1];
  const int*   edge_index = (const int*)d_in[2];
  const float* edge_attr  = (const float*)d_in[4];
  const unsigned char* edge_mask = (const unsigned char*)d_in[5];
  const float* edge_len   = (const float*)d_in[6];
  const float* W1 = (const float*)d_in[8];
  const float* b1 = (const float*)d_in[9];
  const float* W2 = (const float*)d_in[10];
  const float* b2 = (const float*)d_in[11];
  const float* W3 = (const float*)d_in[12];
  const float* b3 = (const float*)d_in[13];
  const float* W4 = (const float*)d_in[14];
  const float* b4 = (const float*)d_in[15];
  const float* inW   = (const float*)d_in[16];
  const float* in_b  = (const float*)d_in[17];
  const float* qkvW  = (const float*)d_in[18];
  const float* qkv_b = (const float*)d_in[19];
  const float* oW    = (const float*)d_in[20];
  const float* o_b   = (const float*)d_in[21];
  const float* ln_g  = (const float*)d_in[22];
  const float* ln_b  = (const float*)d_in[23];
  const float* outW  = (const float*)d_in[24];
  const float* outb  = (const float*)d_in[25];
  float* out = (float*)d_out;

  unsigned char* ws = (unsigned char*)d_ws;
  size_t off = 0;
  auto carve = [&](size_t bytes) {
    unsigned char* p = ws + off;
    off += (bytes + 255) & ~(size_t)255;
    return p;
  };
  __bf16* W1b   = (__bf16*)carve((size_t)4 * 256 * 288 * 2);
  __bf16* W2b   = (__bf16*)carve((size_t)4 * 128 * 256 * 2);
  __bf16* W3b   = (__bf16*)carve((size_t)4 * 64 * 128 * 2);
  __bf16* inWb  = (__bf16*)carve((size_t)128 * 160 * 2);
  __bf16* qkvWb = (__bf16*)carve((size_t)6 * 128 * 128 * 2);
  __bf16* oWb   = (__bf16*)carve((size_t)2 * 128 * 128 * 2);
  float* edge_inv = (float*)carve((size_t)4 * N_EDGES * 4);
  float* node_eq  = (float*)carve((size_t)4 * N_NODES * 3 * 4);
  float* acc_i    = (float*)carve((size_t)4 * N_NODES * 4);
  float* cnt_i    = (float*)carve((size_t)4 * N_NODES * 4);
  float* acc_j    = (float*)carve((size_t)4 * N_NODES * 4);
  float* cnt_j    = (float*)carve((size_t)4 * N_NODES * 4);
  float* logits   = (float*)carve((size_t)N_NODES * 4 * 4);

  // 1) pack weights to bf16 transposed panels
  pack_transpose_kernel<<<(4 * 256 * 288 + 255) / 256, 256, 0, stream>>>(W1, W1b, 4, 257, 256, 288);
  pack_transpose_kernel<<<(4 * 128 * 256 + 255) / 256, 256, 0, stream>>>(W2, W2b, 4, 256, 128, 256);
  pack_transpose_kernel<<<(4 * 64 * 128 + 255) / 256, 256, 0, stream>>>(W3, W3b, 4, 128, 64, 128);
  pack_transpose_kernel<<<(128 * 160 + 255) / 256, 256, 0, stream>>>(inW, inWb, 1, 130, 128, 160);
  pack_transpose_kernel<<<(6 * 128 * 128 + 255) / 256, 256, 0, stream>>>(qkvW, qkvWb, 6, 128, 128, 128);
  pack_transpose_kernel<<<(2 * 128 * 128 + 255) / 256, 256, 0, stream>>>(oW, oWb, 2, 128, 128, 128);

  // 2) zero accumulators (node_eq .. cnt_j are contiguous, sizes 256-aligned)
  size_t zero_bytes = (size_t)4 * N_NODES * 3 * 4 + (size_t)4 * 4 * N_NODES * 4;
  hipMemsetAsync(node_eq, 0, zero_bytes, stream);

  // 3) edge MLP (WMMA + TDM weight staging): one type per block, 2 waves/block
  edge_mlp_kernel<<<8192, 64, 0, stream>>>(h, edge_index, edge_attr,
                                           W1b, b1, W2b, b2, W3b, b3, W4, b4, edge_inv);

  // 4) scatter
  scatter_kernel<<<(N_EDGES + 255) / 256, 256, 0, stream>>>(
      edge_index, coord, edge_attr, edge_len, edge_mask, edge_inv,
      node_eq, acc_i, cnt_i, acc_j, cnt_j);

  // 5) attention (WMMA): 4 nodes/wave, 2 waves/block
  attn_kernel<<<N_NODES / 8, 64, 0, stream>>>(h, acc_i, cnt_i, acc_j, cnt_j,
                                              inWb, in_b, qkvWb, qkv_b, oWb, o_b,
                                              ln_g, ln_b, outW, outb, logits);

  // 6) combine + write outputs
  final_kernel<<<(N_NODES + 255) / 256, 256, 0, stream>>>(coord, logits, node_eq, out);
}
